// VJEPA2RopeAttention_12876311954019
// MI455X (gfx1250) — matrix operand order repaired and checked by reading, not compiled, not run on hardware
//
#include <hip/hip_runtime.h>

typedef _Float16 h16;
typedef __attribute__((ext_vector_type(16))) _Float16 v16h;
typedef __attribute__((ext_vector_type(8)))  _Float16 v8h;
typedef __attribute__((ext_vector_type(8)))  float    v8f;
typedef __attribute__((ext_vector_type(4)))  int      v4i;

#define WMMA(a, b, c) __builtin_amdgcn_wmma_f32_16x16x32_f16(false, (a), false, (b), (short)0, (c), false, false)

#define AS1 __attribute__((address_space(1)))
#define AS3 __attribute__((address_space(3)))

#if defined(__AMDGCN__) && __has_builtin(__builtin_amdgcn_global_load_async_to_lds_b128)
#define HAVE_ASYNC 1
#else
#define HAVE_ASYNC 0
#endif

// 16-byte global -> LDS copy: async (ASYNCcnt) on gfx1250, sync fallback else.
__device__ __forceinline__ void async_cp16(h16* lds, const h16* g) {
#if HAVE_ASYNC
    __builtin_amdgcn_global_load_async_to_lds_b128((AS1 v4i*)(g), (AS3 v4i*)(lds), 0, 0);
#else
    *(v8h*)lds = *(const v8h*)g;
#endif
}

__device__ __forceinline__ void cp_wait() {
#if HAVE_ASYNC
#if __has_builtin(__builtin_amdgcn_s_wait_asynccnt)
    __builtin_amdgcn_s_wait_asynccnt(0);
#else
    asm volatile("s_wait_asynccnt 0" ::: "memory");
#endif
#endif
}

// Same-wave LDS store->load ordering: DS ops from one wave complete in order
// on CDNA5, so a compiler-level ordering fence is sufficient (no block barrier).
__device__ __forceinline__ void wave_lds_fence() {
    __builtin_amdgcn_wave_barrier();
}

__device__ __forceinline__ v8f v8f_zero() {
    v8f r;
#pragma unroll
    for (int i = 0; i < 8; ++i) r[i] = 0.f;
    return r;
}

// A-fragment (16x32, f16) from row-major LDS tile (CDNA5 wave32 16-bit A layout).
__device__ __forceinline__ v16h load_a_frag(const h16* base, int stride) {
    int l = threadIdx.x & 31;
    int r = l & 15;
    int s = (l >> 4) << 3;
    const h16* p = base + r * stride + s;
    union { v16h v; v8h h[2]; } u;
    u.h[0] = *(const v8h*)(p);
    u.h[1] = *(const v8h*)(p + 16);
    return u.v;
}

// B-fragment (32x16, f16): lane L holds column N=L%16; elements s<8 are K=8*(L/16)+s, s>=8 are K=16+8*(L/16)+(s-8).
__device__ __forceinline__ v16h load_b_frag(const h16* base, int stride) {
    int l = threadIdx.x & 31;
    int n = l & 15;
    int hb = (l >> 4) << 3;
    v16h v;
#pragma unroll
    for (int s = 0; s < 8; ++s) {
        v[s] = base[(hb + s) * stride + n];
        v[s + 8] = base[(16 + hb + s) * stride + n];
    }
    return v;
}

// ---------------------------------------------------------------- convert ---
__global__ __launch_bounds__(256) void cvt_f32_f16(const float* __restrict__ in,
                                                   h16* __restrict__ out, size_t n) {
    size_t i = (size_t)blockIdx.x * blockDim.x + threadIdx.x;
    if (i < n) out[i] = (h16)in[i];
}

// ------------------------------------------------------------------- GEMM ---
// C[M,N] = A[M,K] (f16 row-major) * B[K,N] (f16 row-major) + bias[N]
// Block 128x128, TK=32, 8 waves (2x4), 64x32 per wave; double-buffered async LDS.
template <bool OUT_F16>
__global__ __launch_bounds__(256) void gemm_f16(const h16* __restrict__ A,
                                                const h16* __restrict__ Bm,
                                                const float* __restrict__ bias,
                                                void* __restrict__ Cout,
                                                int M, int N, int K) {
    const int TK = 32;
    const int ASTR = 40;    // 32 + 8 pad
    const int BSTR = 136;   // 128 + 8 pad
    __shared__ h16 As[2][128 * 40];
    __shared__ h16 Bs[2][32 * 136];

    int m0 = blockIdx.y * 128, n0 = blockIdx.x * 128;
    int t = threadIdx.x;
    int wave = t >> 5;
    int wm = (wave >> 2) * 64;
    int wn = (wave & 3) * 32;

    int ar = t >> 1, ac = (t & 1) << 4;   // A tile: 128 x 32
    int br = t >> 3, bc = (t & 7) << 4;   // B tile: 32 x 128

    v8f acc[4][2];
#pragma unroll
    for (int i = 0; i < 4; ++i)
#pragma unroll
        for (int j = 0; j < 2; ++j) acc[i][j] = v8f_zero();

    auto stage = [&](int buf, int k0) {
        const h16* ga = A + (size_t)(m0 + ar) * K + k0 + ac;
        async_cp16(&As[buf][ar * ASTR + ac], ga);
        async_cp16(&As[buf][ar * ASTR + ac + 8], ga + 8);
        const h16* gb = Bm + (size_t)(k0 + br) * N + n0 + bc;
        async_cp16(&Bs[buf][br * BSTR + bc], gb);
        async_cp16(&Bs[buf][br * BSTR + bc + 8], gb + 8);
    };

    stage(0, 0);
    cp_wait();
    __syncthreads();

    int cur = 0;
    for (int k0 = 0; k0 < K; k0 += TK) {
        if (k0 + TK < K) stage(cur ^ 1, k0 + TK);
        if (k0 + 2 * TK < K) {  // warm L2 ahead of the async engine
            __builtin_prefetch(A + (size_t)(m0 + ar) * K + k0 + 2 * TK + ac, 0, 1);
            __builtin_prefetch(Bm + (size_t)(k0 + 2 * TK + br) * N + n0 + bc, 0, 1);
        }
        v16h a[4], b[2];
#pragma unroll
        for (int i = 0; i < 4; ++i) a[i] = load_a_frag(&As[cur][(wm + i * 16) * ASTR], ASTR);
#pragma unroll
        for (int j = 0; j < 2; ++j) b[j] = load_b_frag(&Bs[cur][wn + j * 16], BSTR);
#pragma unroll
        for (int i = 0; i < 4; ++i)
#pragma unroll
            for (int j = 0; j < 2; ++j) acc[i][j] = WMMA(a[i], b[j], acc[i][j]);
        cp_wait();
        __syncthreads();
        cur ^= 1;
    }

    int l = t & 31;
    int cn = l & 15;
    int rh = (l >> 4) << 3;
#pragma unroll
    for (int i = 0; i < 4; ++i)
#pragma unroll
        for (int j = 0; j < 2; ++j) {
            int col = n0 + wn + j * 16 + cn;
            float bv = bias ? bias[col] : 0.f;
#pragma unroll
            for (int g = 0; g < 8; ++g) {
                size_t row = (size_t)(m0 + wm + i * 16 + g + rh);
                float v = acc[i][j][g] + bv;
                if (OUT_F16) ((h16*)Cout)[row * N + col] = (h16)v;
                else         ((float*)Cout)[row * N + col] = v;
            }
        }
}

// --------------------------------------------------------- RoPE + permute ---
// In:  Qf/Kf/Vf flat (B,N,D) f16.  Out: Qh (BH,N,64), KhT (BH,64,N), Vh (BH,N,64)
// RoPE per reference: 3 parts of 20 dims (f,h,w positions), freq idx = j % 10.
__global__ __launch_bounds__(256) void rope_permute(const h16* __restrict__ Qf,
                                                    const h16* __restrict__ Kf,
                                                    const h16* __restrict__ Vf,
                                                    h16* __restrict__ Qh,
                                                    h16* __restrict__ KhT,
                                                    h16* __restrict__ Vh) {
    const int N = 2048, D = 1024;
    int t = blockIdx.x * blockDim.x + threadIdx.x;   // (bh, n)
    int n = t & (N - 1);
    int bh = t >> 11;                                 // b*16 + h
    int b = bh >> 4, h = bh & 15;

    int f = n >> 8;
    int rem = n & 255;
    int hh = rem >> 4, wwp = rem & 15;
    float pos[3] = {(float)f, (float)hh, (float)wwp};

    const size_t src = (size_t)(b * N + n) * D + h * 64;
    const h16* q = Qf + src;
    const h16* k = Kf + src;
    h16* qo = Qh + ((size_t)bh * N + n) * 64;
    h16* ko = KhT + (size_t)bh * 64 * N + n;   // stride N over d
    const float L = 0.921034037197618f;        // ln(10000)/10

#pragma unroll
    for (int part = 0; part < 3; ++part) {
        int s = part * 20;
        float p = pos[part];
#pragma unroll
        for (int i = 0; i < 10; ++i) {
            int j0 = 2 * i, j1 = 2 * i + 1;
            float th0 = p * __expf(-L * (float)(j0 % 10));
            float th1 = p * __expf(-L * (float)(j1 % 10));
            float c0 = __cosf(th0), s0 = __sinf(th0);
            float c1 = __cosf(th1), s1 = __sinf(th1);
            float qx0 = (float)q[s + j0], qx1 = (float)q[s + j1];
            float kx0 = (float)k[s + j0], kx1 = (float)k[s + j1];
            qo[s + j0] = (h16)(qx0 * c0 - qx1 * s0);
            qo[s + j1] = (h16)(qx1 * c1 + qx0 * s1);
            ko[(size_t)(s + j0) * N] = (h16)(kx0 * c0 - kx1 * s0);
            ko[(size_t)(s + j1) * N] = (h16)(kx1 * c1 + kx0 * s1);
        }
    }
#pragma unroll
    for (int d = 60; d < 64; ++d) {
        qo[d] = q[d];
        ko[(size_t)d * N] = k[d];
    }
    const h16* v = Vf + src;
    h16* vo = Vh + ((size_t)bh * N + n) * 64;
#pragma unroll
    for (int d = 0; d < 64; d += 8) *(v8h*)(vo + d) = *(const v8h*)(v + d);
}

// -------------------------------------------------------- flash attention ---
// grid (BH=64, N/64=32), 128 threads (4 waves), 16 query rows per wave.
// Double-buffered async K/V tiles; softmax scratch is wave-private LDS.
__global__ __launch_bounds__(128) void attn(const h16* __restrict__ Qh,
                                            const h16* __restrict__ KhT,
                                            const h16* __restrict__ Vh,
                                            h16* __restrict__ ctx) {
    const int N = 2048, HD = 64;
    int bh = blockIdx.x;
    int b = bh >> 4, h = bh & 15;
    int t = threadIdx.x, wave = t >> 5, lane = t & 31;

    __shared__ h16   Kt[2][64 * 72];     // [d][key] (transposed K chunk)
    __shared__ h16   Vt[2][64 * 72];     // [key][d]
    __shared__ float Pf[4][16 * 68];     // raw scores, per wave
    __shared__ h16   Ph[4][16 * 72];     // softmaxed P, per wave
    __shared__ float Alp[4][16];         // per-row rescale / 1/l

    const h16* Qb = Qh + (size_t)bh * N * HD;
    const h16* Kb = KhT + (size_t)bh * HD * N;
    const h16* Vb = Vh + (size_t)bh * N * HD;

    int q0 = blockIdx.y * 64 + wave * 16;

    v16h qa[2];
    {
        int r = lane & 15, s = (lane >> 4) << 3;
        const h16* p = Qb + (size_t)(q0 + r) * HD + s;
#pragma unroll
        for (int kh = 0; kh < 2; ++kh) {
            union { v16h v; v8h h2[2]; } u;
            u.h2[0] = *(const v8h*)(p + kh * 32);
            u.h2[1] = *(const v8h*)(p + kh * 32 + 16);
            qa[kh] = u.v;
        }
    }

    v8f o[4];
#pragma unroll
    for (int j = 0; j < 4; ++j) o[j] = v8f_zero();
    float m_run = -1e30f, l_run = 0.f;
    const float scale = 0.125f;  // 64^-0.5

    int lr = t >> 1, lc = (t & 1) << 5;   // tile loader: 64 rows x 64 cols
    int cn = lane & 15, rh = (lane >> 4) << 3;

    auto stageKV = [&](int buf, int kc) {
        const h16* gk = Kb + (size_t)lr * N + kc + lc;
        h16* dk = &Kt[buf][lr * 72 + lc];
        async_cp16(dk,      gk);      async_cp16(dk + 8,  gk + 8);
        async_cp16(dk + 16, gk + 16); async_cp16(dk + 24, gk + 24);
        const h16* gv = Vb + (size_t)(kc + lr) * HD + lc;
        h16* dv = &Vt[buf][lr * 72 + lc];
        async_cp16(dv,      gv);      async_cp16(dv + 8,  gv + 8);
        async_cp16(dv + 16, gv + 16); async_cp16(dv + 24, gv + 24);
    };

    stageKV(0, 0);
    cp_wait();
    __syncthreads();

    int cur = 0;
    for (int kc = 0; kc < N; kc += 64) {
        if (kc + 64 < N) stageKV(cur ^ 1, kc + 64);

        // S = Q (16x64) * K^T (64 x 64keys): 4 N-tiles, 2 k-steps each
#pragma unroll
        for (int j = 0; j < 4; ++j) {
            v16h b0 = load_b_frag(&Kt[cur][j * 16], 72);
            v16h b1 = load_b_frag(&Kt[cur][32 * 72 + j * 16], 72);
            v8f c = v8f_zero();
            c = WMMA(qa[0], b0, c);
            c = WMMA(qa[1], b1, c);
#pragma unroll
            for (int g = 0; g < 8; ++g)
                Pf[wave][(g + rh) * 68 + j * 16 + cn] = c[g] * scale;
        }
        wave_lds_fence();   // Pf is wave-private; same-wave DS ops are in order

        if (lane < 16) {    // online softmax, one row per lane
            float mx = m_run;
            for (int i = 0; i < 64; ++i) mx = fmaxf(mx, Pf[wave][lane * 68 + i]);
            float alpha = __expf(m_run - mx);
            float sum = 0.f;
            for (int i = 0; i < 64; ++i) {
                float p = __expf(Pf[wave][lane * 68 + i] - mx);
                sum += p;
                Ph[wave][lane * 72 + i] = (h16)p;
            }
            l_run = l_run * alpha + sum;
            m_run = mx;
            Alp[wave][lane] = alpha;
        }
        wave_lds_fence();   // Ph/Alp wave-private

#pragma unroll
        for (int g = 0; g < 8; ++g) {
            float a0 = Alp[wave][g + rh];
#pragma unroll
            for (int j = 0; j < 4; ++j) o[j][g] *= a0;
        }

        v16h pa[2];
        {
            int r = lane & 15, s = (lane >> 4) << 3;
#pragma unroll
            for (int kh = 0; kh < 2; ++kh) {
                union { v16h v; v8h h2[2]; } u;
                u.h2[0] = *(const v8h*)&Ph[wave][r * 72 + kh * 32 + s];
                u.h2[1] = *(const v8h*)&Ph[wave][r * 72 + kh * 32 + s + 16];
                pa[kh] = u.v;
            }
        }
#pragma unroll
        for (int j = 0; j < 4; ++j) {
            v16h b0 = load_b_frag(&Vt[cur][j * 16], 72);
            v16h b1 = load_b_frag(&Vt[cur][32 * 72 + j * 16], 72);
            o[j] = WMMA(pa[0], b0, o[j]);
            o[j] = WMMA(pa[1], b1, o[j]);
        }

        cp_wait();          // next K/V tile landed
        __syncthreads();    // block-shared tile handoff (the only block barrier)
        cur ^= 1;
    }

    if (lane < 16) Alp[wave][lane] = 1.0f / l_run;
    wave_lds_fence();
#pragma unroll
    for (int g = 0; g < 8; ++g) {
        float linv = Alp[wave][g + rh];
        int row = q0 + g + rh;
        size_t ob = ((size_t)(b * N + row)) * 1024 + h * 64;
#pragma unroll
        for (int j = 0; j < 4; ++j) ctx[ob + j * 16 + cn] = (h16)(o[j][g] * linv);
    }
}

// ----------------------------------------------------------------- launch ---
extern "C" void kernel_launch(void* const* d_in, const int* in_sizes, int n_in,
                              void* d_out, int out_size, void* d_ws, size_t ws_size,
                              hipStream_t stream) {
    (void)in_sizes; (void)n_in; (void)out_size; (void)ws_size;
    const float* X  = (const float*)d_in[0];
    const float* Wq = (const float*)d_in[1];
    const float* bq = (const float*)d_in[2];
    const float* Wk = (const float*)d_in[3];
    const float* bk = (const float*)d_in[4];
    const float* Wv = (const float*)d_in[5];
    const float* bv = (const float*)d_in[6];
    const float* Wo = (const float*)d_in[7];
    const float* bo = (const float*)d_in[8];

    const size_t MB = 1024ull * 1024ull;
    char* ws = (char*)d_ws;
    h16* Xh  = (h16*)(ws);              // 16 MB  (8192x1024)
    h16* Wqh = (h16*)(ws + 16 * MB);    //  2 MB each
    h16* Wkh = (h16*)(ws + 18 * MB);
    h16* Wvh = (h16*)(ws + 20 * MB);
    h16* Woh = (h16*)(ws + 22 * MB);
    h16* Qf  = (h16*)(ws + 24 * MB);    // 16 MB flat QKV (f16)
    h16* Kf  = (h16*)(ws + 40 * MB);
    h16* Vf  = (h16*)(ws + 56 * MB);
    h16* Qhd = (h16*)(ws + 72 * MB);    // head-major
    h16* KhT = (h16*)(ws + 88 * MB);    // head-major transposed
    h16* Vhd = (h16*)(ws + 104 * MB);
    h16* ctx = Qf;                      // reuse Qf region after rope consumed it

    const int M = 8192, Nd = 1024, K = 1024;

    size_t nX = (size_t)M * K;
    cvt_f32_f16<<<(unsigned)((nX + 255) / 256), 256, 0, stream>>>(X, Xh, nX);
    size_t nW = (size_t)K * Nd;
    cvt_f32_f16<<<(unsigned)((nW + 255) / 256), 256, 0, stream>>>(Wq, Wqh, nW);
    cvt_f32_f16<<<(unsigned)((nW + 255) / 256), 256, 0, stream>>>(Wk, Wkh, nW);
    cvt_f32_f16<<<(unsigned)((nW + 255) / 256), 256, 0, stream>>>(Wv, Wvh, nW);
    cvt_f32_f16<<<(unsigned)((nW + 255) / 256), 256, 0, stream>>>(Wo, Woh, nW);

    dim3 gg(Nd / 128, M / 128);
    gemm_f16<true><<<gg, 256, 0, stream>>>(Xh, Wqh, bq, Qf, M, Nd, K);
    gemm_f16<true><<<gg, 256, 0, stream>>>(Xh, Wkh, bk, Kf, M, Nd, K);
    gemm_f16<true><<<gg, 256, 0, stream>>>(Xh, Wvh, bv, Vf, M, Nd, K);

    rope_permute<<<512, 256, 0, stream>>>(Qf, Kf, Vf, Qhd, KhT, Vhd);

    attn<<<dim3(64, 32), 128, 0, stream>>>(Qhd, KhT, Vhd, ctx);

    gemm_f16<false><<<gg, 256, 0, stream>>>(ctx, Woh, bo, d_out, M, Nd, K);
}
